// FeedBack_65455301591861
// MI455X (gfx1250) — matrix operand order repaired and checked by reading, not compiled, not run on hardware
//
#include <hip/hip_runtime.h>

#define B_TOTAL   16384
#define T_IN      48
#define F_DIM     4
#define UNITS     256
#define OUT_STEPS 24
#define KTOT      288      // 256 (h) + 4 (x) + 28 zero pad -> 9 WMMA K-chunks of 32
#define A_STRIDE  304      // padded LDS row stride in bf16 elements
#define M_TILE    32       // batch rows per workgroup
#define GSTRIDE   (UNITS * KTOT * 2)   // byte stride between gate blocks in WcT = 147456

typedef __attribute__((ext_vector_type(16))) __bf16       bf16x16;
typedef __attribute__((ext_vector_type(8)))  float        f32x8;
typedef __attribute__((ext_vector_type(4)))  unsigned int u32x4;

// global (addrspace 1) types so B loads lower to global_load_b128 with
// saddr(uniform base) + voffset(lane) + imm(g,kk) addressing.
typedef __attribute__((address_space(1))) const char   gchar;
typedef __attribute__((address_space(1))) const u32x4  gcu32x4;

union AFrag { bf16x16 v; u32x4 u[2]; };

__device__ __forceinline__ unsigned short f2bf(float f) {
  unsigned int u = __float_as_uint(f);
  u += 0x7FFFu + ((u >> 16) & 1u);          // round-to-nearest-even
  return (unsigned short)(u >> 16);
}
__device__ __forceinline__ float bf2f(unsigned short s) {
  return __uint_as_float(((unsigned int)s) << 16);
}
__device__ __forceinline__ float sigm(float x)  { return 1.0f / (1.0f + __expf(-x)); }
__device__ __forceinline__ float tanha(float x) { return 2.0f / (1.0f + __expf(-2.0f * x)) - 1.0f; }

// One 16x16(K=32) B fragment: two runs of 8 bf16 (k and k+16) per lane = two global b128.
__device__ __forceinline__ void loadB(AFrag& b, gchar* p) {
  b.u[0] = *(const gcu32x4*)p;
  b.u[1] = *(const gcu32x4*)(p + 32);
}
__device__ __forceinline__ void loadA(AFrag& a, const unsigned short* ap) {
  a.u[0] = *(const u32x4*)ap;
  a.u[1] = *(const u32x4*)(ap + 16);
}

// Pack [Wh^T | Wx^T | 0] into a per-unit-contiguous bf16 matrix: WcT[n][k], k in [0,288)
__global__ void lstm_pack_weights(const float* __restrict__ Wx,
                                  const float* __restrict__ Wh,
                                  unsigned short* __restrict__ WcT) {
  const int n = blockIdx.x;    // 0..1023 (gate column)
  const int k = threadIdx.x;   // 0..287
  float v = 0.0f;
  if (k < UNITS)               v = Wh[k * (4 * UNITS) + n];
  else if (k < UNITS + F_DIM)  v = Wx[(k - UNITS) * (4 * UNITS) + n];
  WcT[n * KTOT + k] = f2bf(v);
}

__global__ __launch_bounds__(256, 1)
void lstm_feedback(const float* __restrict__ x,
                   const float* __restrict__ bias,
                   const float* __restrict__ Wd,
                   const float* __restrict__ bd,
                   const unsigned short* __restrict__ WcT,
                   float* __restrict__ out) {
  __shared__ __align__(16) unsigned short Abuf[2][M_TILE * A_STRIDE]; // bf16 A operand, double buffered
  __shared__ float WdT[F_DIM * UNITS];   // Wd transposed: WdT[f][u]

  const int tid  = threadIdx.x;
  const int wave = tid >> 5;
  const int lane = tid & 31;
  const int half = lane >> 4;
  const int l16  = lane & 15;
  const int r0   = blockIdx.x * M_TILE;

  for (int i = tid; i < 2 * M_TILE * A_STRIDE; i += 256) ((unsigned short*)Abuf)[i] = 0;
  for (int i = tid; i < F_DIM * UNITS; i += 256) {
    const int f = i >> 8, u = i & (UNITS - 1);
    WdT[i] = Wd[u * F_DIM + f];
  }

  // Per-lane bias: C-layout fixes the column per lane -> scalar per (gate, subtile)
  float biasr[4][2];
#pragma unroll
  for (int g = 0; g < 4; ++g)
#pragma unroll
    for (int s = 0; s < 2; ++s)
      biasr[g][s] = bias[g * UNITS + wave * 32 + s * 16 + l16];

  const f32x8 zero8 = {0.f, 0.f, 0.f, 0.f, 0.f, 0.f, 0.f, 0.f};
  f32x8 cst[2][2];                       // cell state in registers (accumulator layout)
#pragma unroll
  for (int mt = 0; mt < 2; ++mt)
#pragma unroll
    for (int s = 0; s < 2; ++s) cst[mt][s] = zero8;

  unsigned long long wbase = (unsigned long long)WcT;

  __syncthreads();

  for (int t = 0; t < T_IN + OUT_STEPS - 1; ++t) {   // 71 steps
    const int cur = t & 1, nxt = cur ^ 1;

    // warmup input -> x-slot (cols 256..259) of current A buffer
    if (t < T_IN && tid < M_TILE * F_DIM) {
      const int r = tid >> 2, f = tid & 3;
      Abuf[cur][r * A_STRIDE + UNITS + f] =
          f2bf(x[(size_t)(r0 + r) * T_IN * F_DIM + t * F_DIM + f]);
    }
    __syncthreads();

#pragma unroll
    for (int mt = 0; mt < 2; ++mt) {
#pragma unroll
      for (int s = 0; s < 2; ++s) {
        // Launder the weight base: keeps B loads formally loop-variant so LICM
        // can't hoist ~576 VGPRs of weights across the time loop (spill storm).
        asm volatile("" : "+s"(wbase));

        f32x8 acc0 = zero8, acc1 = zero8, acc2 = zero8, acc3 = zero8;
        const int colBase = wave * 32 + s * 16 + l16;   // unit column for this lane
        const int rowOff  = (mt * 16 + l16) * A_STRIDE;
        const int kaH     = 8 * half;                   // A K-offset within chunk

        // uniform base (SGPR) + divergent 32-bit lane offset (one VGPR);
        // per-(g,kk) offsets are compile-time immediates folded into the load.
        gchar* wlane = (gchar*)wbase + (colBase * KTOT + 8 * half) * 2;

        // software pipeline, prefetch distance 2 (triple-buffered fragments);
        // sched_barriers pin the issue order so the scheduler can't sink the
        // prefetch loads back to just-in-time (seen in the previous build).
        AFrag a[3], bb[3][4];
        loadA(a[0], &Abuf[cur][rowOff + kaH]);
#pragma unroll
        for (int g = 0; g < 4; ++g) loadB(bb[0][g], wlane + g * GSTRIDE);
        loadA(a[1], &Abuf[cur][rowOff + 32 + kaH]);
#pragma unroll
        for (int g = 0; g < 4; ++g) loadB(bb[1][g], wlane + g * GSTRIDE + 64);

#pragma unroll
        for (int kk = 0; kk < 9; ++kk) {
          const int cu_ = kk % 3;
          if (kk < 7) {
            const int nx_ = (kk + 2) % 3;
            loadA(a[nx_], &Abuf[cur][rowOff + (kk + 2) * 32 + kaH]);
#pragma unroll
            for (int g = 0; g < 4; ++g)
              loadB(bb[nx_][g], wlane + g * GSTRIDE + (kk + 2) * 64);
          }
          __builtin_amdgcn_sched_barrier(0);   // loads(kk+2) stay above wmmas(kk)
          acc0 = __builtin_amdgcn_wmma_f32_16x16x32_bf16(false, a[cu_].v, false, bb[cu_][0].v, (short)0, acc0, false, false);
          acc1 = __builtin_amdgcn_wmma_f32_16x16x32_bf16(false, a[cu_].v, false, bb[cu_][1].v, (short)0, acc1, false, false);
          acc2 = __builtin_amdgcn_wmma_f32_16x16x32_bf16(false, a[cu_].v, false, bb[cu_][2].v, (short)0, acc2, false, false);
          acc3 = __builtin_amdgcn_wmma_f32_16x16x32_bf16(false, a[cu_].v, false, bb[cu_][3].v, (short)0, acc3, false, false);
          __builtin_amdgcn_sched_barrier(0);   // wmmas(kk) stay above loads(kk+3)
        }

        // i,f,g,o for the same (row, unit) sit in the same lane/VGPR slot -> gates in registers
#pragma unroll
        for (int e = 0; e < 8; ++e) {
          const float iv = sigm (acc0[e] + biasr[0][s]);
          const float fv = sigm (acc1[e] + biasr[1][s]);
          const float gv = tanha(acc2[e] + biasr[2][s]);
          const float ov = sigm (acc3[e] + biasr[3][s]);
          const float cn = fv * cst[mt][s][e] + iv * gv;
          cst[mt][s][e]  = cn;
          const float hv = ov * tanha(cn);
          const int row  = mt * 16 + e + 8 * half;      // C-layout row for element e
          Abuf[nxt][row * A_STRIDE + colBase] = f2bf(hv);
        }
      }
    }
    __syncthreads();

    // decoder: pred = h @ Wd + bd ; emit output and feed back as next input
    if (t >= T_IN - 1 && tid < M_TILE * F_DIM) {
      const int r = tid >> 2, f = tid & 3;
      float acc = bd[f];
      const u32x4* hp = (const u32x4*)&Abuf[nxt][r * A_STRIDE];  // 8 bf16 per b128
#pragma unroll
      for (int uu = 0; uu < UNITS / 8; ++uu) {
        const u32x4 hv = hp[uu];
#pragma unroll
        for (int j = 0; j < 4; ++j) {
          acc += bf2f((unsigned short)(hv[j] & 0xFFFFu)) * WdT[f * UNITS + uu * 8 + 2 * j];
          acc += bf2f((unsigned short)(hv[j] >> 16))     * WdT[f * UNITS + uu * 8 + 2 * j + 1];
        }
      }
      out[(size_t)(r0 + r) * OUT_STEPS * F_DIM + (t - (T_IN - 1)) * F_DIM + f] = acc;
      if (t < T_IN + OUT_STEPS - 2)
        Abuf[nxt][r * A_STRIDE + UNITS + f] = f2bf(acc);
    }
    // loop-top __syncthreads covers pred/x-slot writes before next step's reads
  }
}

extern "C" void kernel_launch(void* const* d_in, const int* in_sizes, int n_in,
                              void* d_out, int out_size, void* d_ws, size_t ws_size,
                              hipStream_t stream) {
  const float* x  = (const float*)d_in[0];
  const float* Wx = (const float*)d_in[1];
  const float* Wh = (const float*)d_in[2];
  const float* b  = (const float*)d_in[3];
  const float* Wd = (const float*)d_in[4];
  const float* bd = (const float*)d_in[5];
  float* out = (float*)d_out;
  unsigned short* WcT = (unsigned short*)d_ws;   // 1024 * 288 * 2B = 576 KB

  lstm_pack_weights<<<4 * UNITS, KTOT, 0, stream>>>(Wx, Wh, WcT);
  lstm_feedback<<<B_TOTAL / M_TILE, 256, 0, stream>>>(x, b, Wd, bd, WcT, out);
}
